// SpatialAttention_68092411511608
// MI455X (gfx1250) — compile-verified, matrix operand order checked by reading, not run on hardware
//
#include <hip/hip_runtime.h>

typedef __attribute__((ext_vector_type(16))) _Float16 v16h;
typedef __attribute__((ext_vector_type(8)))  _Float16 v8h;
typedef __attribute__((ext_vector_type(8)))  float    v8f;
typedef __attribute__((ext_vector_type(2)))  _Float16 h2;

#define HEADS 4
#define DH    32
#define DIM   128
#define NTOK  4096            // 64*64
#define BFRM  4               // b*f
#define ROWS  16384           // BFRM*NTOK

// K index of element 2v (element 2v+1 = +1) of a 16-bit A/B WMMA fragment
// per CDNA5 ISA 7.12.2: lanes 0-15 hold K {0..7,16..23}, lanes 16-31 hold {8..15,24..31}
__device__ __forceinline__ int akpair(int lane, int v) {
  return ((v >> 2) << 4) + ((lane >> 4) << 3) + ((v & 3) << 1);
}

// Build a 16-bit A/B fragment from a 32-element contiguous row: this lane's
// halves are rows[8h .. 8h+7] and rows[16+8h .. 16+8h+7] -> two b128 loads.
__device__ __forceinline__ v16h ldfrag(const _Float16* __restrict__ row, int half) {
  v8h lo = *(const v8h*)(row + (half << 3));
  v8h hi = *(const v8h*)(row + 16 + (half << 3));
  return __builtin_shufflevector(lo, hi, 0, 1, 2, 3, 4, 5, 6, 7,
                                 8, 9, 10, 11, 12, 13, 14, 15);
}

__device__ __forceinline__ float fast_exp2(float x) {
#if __has_builtin(__builtin_amdgcn_exp2f)
  return __builtin_amdgcn_exp2f(x);   // v_exp_f32
#else
  return exp2f(x);
#endif
}

// ---------------- Kernel 1: QKV projection -----------------------------------
// Q stored pre-scaled by dh^-0.5 * log2(e) (softmax done in exp2 domain),
// K as [bh][n][d] f16, V transposed [bh][d][n] f16.
__global__ void qkv_kernel(const float* __restrict__ x, const float* __restrict__ w_qkv,
                           _Float16* __restrict__ qh, _Float16* __restrict__ kh,
                           _Float16* __restrict__ vth) {
  const int lane = threadIdx.x;
  const int m = lane & 15, half = lane >> 4;
  const int row0 = blockIdx.x << 4;   // [0,16384)
  const int col0 = blockIdx.y << 4;   // [0,384)
  const int seg  = blockIdx.y >> 3;   // 0=Q, 1=K, 2=V (scalar branch)
  v8f acc = {};
  for (int kb = 0; kb < DIM; kb += 32) {
    v16h a, b;
#pragma unroll
    for (int v = 0; v < 8; ++v) {
      int k = kb + akpair(lane, v);
      a[2 * v]     = (_Float16)x[(size_t)k * ROWS + row0 + m];
      a[2 * v + 1] = (_Float16)x[(size_t)(k + 1) * ROWS + row0 + m];
      b[2 * v]     = (_Float16)w_qkv[k * 384 + col0 + m];
      b[2 * v + 1] = (_Float16)w_qkv[(k + 1) * 384 + col0 + m];
    }
    acc = __builtin_amdgcn_wmma_f32_16x16x32_f16(false, a, false, b, (short)0, acc, false, false);
  }
  const float qscale = 0.17677669529663687f * 1.4426950408889634f; // dh^-0.5 * log2e
  const int cs = ((blockIdx.y & 7) << 4) + m; // col within segment, 0..127
  const int head = cs >> 5, d = cs & 31;
  if (seg == 0) {
#pragma unroll
    for (int r = 0; r < 8; ++r) {
      int grow = row0 + r + (half << 3);
      int bf = grow >> 12, n = grow & (NTOK - 1);
      qh[(size_t)(((bf << 2) + head) * NTOK + n) * DH + d] = (_Float16)(acc[r] * qscale);
    }
  } else if (seg == 1) {
#pragma unroll
    for (int r = 0; r < 8; ++r) {
      int grow = row0 + r + (half << 3);
      int bf = grow >> 12, n = grow & (NTOK - 1);
      kh[(size_t)(((bf << 2) + head) * NTOK + n) * DH + d] = (_Float16)acc[r];
    }
  } else {
#pragma unroll
    for (int r = 0; r < 8; ++r) {
      int grow = row0 + r + (half << 3);
      int bf = grow >> 12, n = grow & (NTOK - 1);
      vth[(size_t)(((bf << 2) + head) * DH + d) * NTOK + n] = (_Float16)acc[r];
    }
  }
}

// ---------------- Kernel 2: flash attention (transposed-S, 32 queries/wave) --
__global__ void attn_kernel(const _Float16* __restrict__ qh, const _Float16* __restrict__ kh,
                            const _Float16* __restrict__ vth, _Float16* __restrict__ ao) {
  const int lane = threadIdx.x;
  const int m = lane & 15, half = lane >> 4;
  const int bh = blockIdx.x;                // bf*4+head
  const int qt = blockIdx.y;                // 0..127 (32 queries each)
  const _Float16* Qp = qh + (size_t)(bh * NTOK + (qt << 5)) * DH;
  const _Float16* Kp = kh + (size_t)bh * NTOK * DH;
  const _Float16* Vp = vth + (size_t)bh * DH * NTOK;

  // Q^T as B-fragments (lane col = query m / 16+m)
  v16h qb0 = ldfrag(Qp + m * DH, half);
  v16h qb1 = ldfrag(Qp + (16 + m) * DH, half);

  v8f o00 = {}, o01 = {}, o10 = {}, o11 = {};   // O^T: [tile][d-half]
  float mc0 = -3.0e38f, mc1 = -3.0e38f, ls0 = 0.f, ls1 = 0.f;

  for (int j = 0; j < NTOK; j += 32) {
    v16h ka0 = ldfrag(Kp + (j + m) * DH, half);        // keys j+m rows
    v16h ka1 = ldfrag(Kp + (j + 16 + m) * DH, half);
    v8f z = {};
    v8f s00 = __builtin_amdgcn_wmma_f32_16x16x32_f16(false, ka0, false, qb0, (short)0, z, false, false);
    v8f s10 = __builtin_amdgcn_wmma_f32_16x16x32_f16(false, ka1, false, qb0, (short)0, z, false, false);
    v8f s01 = __builtin_amdgcn_wmma_f32_16x16x32_f16(false, ka0, false, qb1, (short)0, z, false, false);
    v8f s11 = __builtin_amdgcn_wmma_f32_16x16x32_f16(false, ka1, false, qb1, (short)0, z, false, false);

    v16h va0 = ldfrag(Vp + m * NTOK + j, half);        // V^T rows d=m / 16+m
    v16h va1 = ldfrag(Vp + (16 + m) * NTOK + j, half);
    {
      int jn = (j + 32) & (NTOK - 1);
      __builtin_prefetch(Kp + (jn + m) * DH, 0, 0);    // -> global_prefetch_b8
      __builtin_prefetch(Vp + m * NTOK + jn, 0, 0);
    }

    // Per-lane block max (all 16 values in a lane share one query row);
    // partner (lane^16) covers the other half of the keys of the same row.
    float mx0 = fmaxf(s00[0], s10[0]);
    float mx1 = fmaxf(s01[0], s11[0]);
#pragma unroll
    for (int r = 1; r < 8; ++r) {
      mx0 = fmaxf(mx0, fmaxf(s00[r], s10[r]));
      mx1 = fmaxf(mx1, fmaxf(s01[r], s11[r]));
    }
    mx0 = fmaxf(mx0, __shfl_xor(mx0, 16, 32));
    mx1 = fmaxf(mx1, __shfl_xor(mx1, 16, 32));

    // Wave-uniform skip of accumulator rescale when no row's max moved.
    if (__ballot((mx0 > mc0) || (mx1 > mc1)) != 0ull) {
      float mn0 = fmaxf(mc0, mx0), mn1 = fmaxf(mc1, mx1);
      float c0 = fast_exp2(mc0 - mn0), c1 = fast_exp2(mc1 - mn1);
      mc0 = mn0; mc1 = mn1;
      ls0 *= c0; ls1 *= c1;
#pragma unroll
      for (int r = 0; r < 8; ++r) {
        o00[r] *= c0; o01[r] *= c0;
        o10[r] *= c1; o11[r] *= c1;
      }
    }

    // P^T B-fragments: C-frag of S^T maps element-for-element onto B layout.
    v16h pb0, pb1;
    float ps0 = 0.f, ps1 = 0.f;
#pragma unroll
    for (int r = 0; r < 8; ++r) {
      float a0 = fast_exp2(s00[r] - mc0), b0 = fast_exp2(s10[r] - mc0);
      float a1 = fast_exp2(s01[r] - mc1), b1 = fast_exp2(s11[r] - mc1);
      ps0 += a0 + b0; ps1 += a1 + b1;
      pb0[r] = (_Float16)a0; pb0[8 + r] = (_Float16)b0;
      pb1[r] = (_Float16)a1; pb1[8 + r] = (_Float16)b1;
    }
    ls0 += ps0; ls1 += ps1;

    o00 = __builtin_amdgcn_wmma_f32_16x16x32_f16(false, va0, false, pb0, (short)0, o00, false, false);
    o01 = __builtin_amdgcn_wmma_f32_16x16x32_f16(false, va1, false, pb0, (short)0, o01, false, false);
    o10 = __builtin_amdgcn_wmma_f32_16x16x32_f16(false, va0, false, pb1, (short)0, o10, false, false);
    o11 = __builtin_amdgcn_wmma_f32_16x16x32_f16(false, va1, false, pb1, (short)0, o11, false, false);
  }

  float inv0 = 1.0f / (ls0 + __shfl_xor(ls0, 16, 32));
  float inv1 = 1.0f / (ls1 + __shfl_xor(ls1, 16, 32));

  int bf = bh >> 2, head = bh & 3;
  _Float16* dst0 = ao + (size_t)(bf * NTOK + (qt << 5) + m) * DIM + head * DH;
  _Float16* dst1 = ao + (size_t)(bf * NTOK + (qt << 5) + 16 + m) * DIM + head * DH;
#pragma unroll
  for (int i = 0; i < 4; ++i) {               // O^T rows are d; pack pairs
    h2 w;
    w.x = (_Float16)(o00[2 * i] * inv0); w.y = (_Float16)(o00[2 * i + 1] * inv0);
    *(h2*)(dst0 + (half << 3) + 2 * i) = w;
    w.x = (_Float16)(o01[2 * i] * inv0); w.y = (_Float16)(o01[2 * i + 1] * inv0);
    *(h2*)(dst0 + 16 + (half << 3) + 2 * i) = w;
    w.x = (_Float16)(o10[2 * i] * inv1); w.y = (_Float16)(o10[2 * i + 1] * inv1);
    *(h2*)(dst1 + (half << 3) + 2 * i) = w;
    w.x = (_Float16)(o11[2 * i] * inv1); w.y = (_Float16)(o11[2 * i + 1] * inv1);
    *(h2*)(dst1 + 16 + (half << 3) + 2 * i) = w;
  }
}

// ---------------- Kernel 3: output projection + scatter ----------------------
__global__ void proj_kernel(const _Float16* __restrict__ ao, const float* __restrict__ w_out,
                            float* __restrict__ out) {
  const int lane = threadIdx.x;
  const int m = lane & 15, half = lane >> 4;
  const int row0 = blockIdx.x << 4;   // [0,16384)
  const int col0 = blockIdx.y << 4;   // [0,128)
  v8f acc = {};
  for (int kb = 0; kb < DIM; kb += 32) {
    v16h a = ldfrag(ao + (size_t)(row0 + m) * DIM + kb, half);
    v16h b;
#pragma unroll
    for (int v = 0; v < 8; ++v) {
      int k = kb + akpair(lane, v);
      b[2 * v]     = (_Float16)w_out[k * DIM + col0 + m];
      b[2 * v + 1] = (_Float16)w_out[(k + 1) * DIM + col0 + m];
    }
    acc = __builtin_amdgcn_wmma_f32_16x16x32_f16(false, a, false, b, (short)0, acc, false, false);
  }
#pragma unroll
  for (int r = 0; r < 8; ++r) {
    int grow = row0 + r + (half << 3);
    int f = grow >> 12, hw = grow & (NTOK - 1);
    int c = col0 + m;
    out[(size_t)c * ROWS + f * NTOK + hw] = acc[r];   // (b,c,f,h,w)
  }
}

extern "C" void kernel_launch(void* const* d_in, const int* in_sizes, int n_in,
                              void* d_out, int out_size, void* d_ws, size_t ws_size,
                              hipStream_t stream) {
  (void)in_sizes; (void)n_in; (void)out_size; (void)ws_size;
  const float* x     = (const float*)d_in[0];
  const float* w_qkv = (const float*)d_in[1];
  const float* w_out = (const float*)d_in[2];
  float* out = (float*)d_out;

  // workspace: Q, K, Vt, attn_out — 4 MB each (f16), 16 MB total
  const size_t seg = (size_t)ROWS * DIM;     // 2M halves
  _Float16* qh  = (_Float16*)d_ws;
  _Float16* kh  = qh  + seg;
  _Float16* vth = kh  + seg;
  _Float16* ao  = vth + seg;

  qkv_kernel <<<dim3(ROWS / 16, 24),           32, 0, stream>>>(x, w_qkv, qh, kh, vth);
  attn_kernel<<<dim3(BFRM * HEADS, NTOK / 32), 32, 0, stream>>>(qh, kh, vth, ao);
  proj_kernel<<<dim3(ROWS / 16, DIM / 16),     32, 0, stream>>>(ao, w_out, out);
}
